// DiagonalWindowAttention_50955492000209
// MI455X (gfx1250) — compile-verified, matrix operand order checked by reading, not run on hardware
//
#include <hip/hip_runtime.h>

#define WH 8
#define WW 8
#define SHF 2
#define NH 6
#define ED 96
#define NP 64
#define S2 4
#define HD 16
#define NWM 64
#define ROWS (NP * S2)      // 256
#define KPAD 72             // padded LDS row stride (halves) for 64-wide tiles
#define WPAD 104            // padded LDS row stride (halves) for 96-wide tiles

typedef __attribute__((ext_vector_type(16))) _Float16 v16h;
typedef __attribute__((ext_vector_type(8)))  _Float16 v8h;
typedef __attribute__((ext_vector_type(4)))  _Float16 v4h;
typedef __attribute__((ext_vector_type(8)))  float    v8f;

// group_partition: patch p=(h,w), shuffle s=(sh,sw) -> source row in (wh*sh*ww*sw)
__device__ __forceinline__ int rowOf(int p, int s) {
    return ((p >> 3) << 5) + ((s >> 1) << 4) + ((p & 7) << 1) + (s & 1);
}

// Load a v16h WMMA A/B fragment chunk from an LDS tile (row-major, `stride` halves per row).
// Per the 16-bit A/B layout: lanes 0-15 hold K = {0..7, 16..23}+0, lanes 16-31 hold +8.
__device__ __forceinline__ v16h ldsFrag(const _Float16* base, int row, int chunk, int off, int stride) {
    const _Float16* p0 = base + row * stride + chunk * 32 + off;
    v8h lo = *(const v8h*)(p0);
    v8h hi = *(const v8h*)(p0 + 16);
    return __builtin_shufflevector(lo, hi, 0,1,2,3,4,5,6,7,8,9,10,11,12,13,14,15);
}

__global__ __launch_bounds__(256)
void dwa_attn_kernel(const float* __restrict__ q_in, const float* __restrict__ k_in,
                     const float* __restrict__ v_in, const float* __restrict__ mask,
                     const float* __restrict__ bias_table,
                     const float* __restrict__ gamma, const float* __restrict__ beta,
                     float* __restrict__ xout)
{
    __shared__ __align__(16) _Float16 q_t[2][NP * KPAD];   // also reused to hold softmax probs
    __shared__ __align__(16) _Float16 k_t[2][NP * KPAD];
    __shared__ __align__(16) _Float16 vT_t[2][NP * KPAD];
    __shared__ float mu_s[NP];
    __shared__ float rs_s[NP];

    const int w    = blockIdx.x;
    const int tid  = threadIdx.x;
    const int lane = tid & 31;
    const int wave = tid >> 5;

    const float* qp_g = q_in + (size_t)w * (ROWS * ED);
    const float* kp_g = k_in + (size_t)w * (ROWS * ED);
    const float* vp_g = v_in + (size_t)w * (ROWS * ED);
    float*       xw   = xout + (size_t)w * (ROWS * ED);
    const float* mk   = mask + (size_t)(w & (NWM - 1)) * (NP * NP);

    // ---- LayerNorm statistics: per patch, over (s2, ed) = 384 values ----
    {
        int p = tid >> 2, s = tid & 3;
        const float* row = qp_g + rowOf(p, s) * ED;
        float sm = 0.f, sq = 0.f;
        for (int i = 0; i < ED; i += 4) {
            float4 x = *(const float4*)(row + i);
            sm += x.x + x.y + x.z + x.w;
            sq += x.x * x.x + x.y * x.y + x.z * x.z + x.w * x.w;
        }
        sm += __shfl_xor(sm, 1, 32);  sq += __shfl_xor(sq, 1, 32);
        sm += __shfl_xor(sm, 2, 32);  sq += __shfl_xor(sq, 2, 32);
        if (s == 0) {
            float m = sm * (1.f / 384.f);
            float v = sq * (1.f / 384.f) - m * m;
            mu_s[p] = m;
            rs_s[p] = rsqrtf(v + 1e-5f);
        }
    }
    __syncthreads();

    const int laneLo = lane & 15;
    const int off    = (lane >> 4) << 3;     // 0 or 8 (fragment K offset per half-wave)
    const int strip  = wave & 3;             // q-row strip (16 rows) this wave owns
    const int grp    = wave >> 2;            // which of the 2 concurrent heads
    const int mrow   = strip * 16 + laneLo;  // A-fragment row (M = lane%16)

    for (int it = 0; it < 3; ++it) {
        // ---- stage two heads' q/k/vT tiles into LDS (f16), float4 granularity ----
        for (int g = 0; g < 2; ++g) {
            int h = it * 2 + g;
            for (int idx = tid; idx < NP * 16; idx += 256) {
                int p  = idx >> 4;
                int c4 = (idx & 15) << 2;        // col base within [0,64), multiple of 4
                int s  = c4 >> 4;
                int r  = rowOf(p, s);
                int gc = h * HD + (c4 & 15);     // global channel base (16B aligned)
                float4 qv = *(const float4*)(qp_g + r * ED + gc);
                float4 kv = *(const float4*)(kp_g + r * ED + gc);
                float4 vv = *(const float4*)(vp_g + r * ED + gc);
                float4 gm = *(const float4*)(gamma + s * ED + gc);
                float4 bt = *(const float4*)(beta  + s * ED + gc);
                float m = mu_s[p], rsd = rs_s[p];
                v4h qh;
                qh[0] = (_Float16)(((qv.x - m) * rsd * gm.x + bt.x) * 0.125f);  // SCALE
                qh[1] = (_Float16)(((qv.y - m) * rsd * gm.y + bt.y) * 0.125f);
                qh[2] = (_Float16)(((qv.z - m) * rsd * gm.z + bt.z) * 0.125f);
                qh[3] = (_Float16)(((qv.w - m) * rsd * gm.w + bt.w) * 0.125f);
                *(v4h*)(&q_t[g][p * KPAD + c4]) = qh;
                v4h kh;
                kh[0] = (_Float16)kv.x;  kh[1] = (_Float16)kv.y;
                kh[2] = (_Float16)kv.z;  kh[3] = (_Float16)kv.w;
                *(v4h*)(&k_t[g][p * KPAD + c4]) = kh;
                vT_t[g][(c4 + 0) * KPAD + p] = (_Float16)vv.x;
                vT_t[g][(c4 + 1) * KPAD + p] = (_Float16)vv.y;
                vT_t[g][(c4 + 2) * KPAD + p] = (_Float16)vv.z;
                vT_t[g][(c4 + 3) * KPAD + p] = (_Float16)vv.w;
            }
        }
        __syncthreads();

        const int head = it * 2 + grp;
        const _Float16* qb = q_t[grp];
        const _Float16* kb = k_t[grp];
        const _Float16* vb = vT_t[grp];

        // ---- attn = q @ k^T : 4 tiles x (K=64 as 2 chained 16x16x32) ----
        v16h a0 = ldsFrag(qb, mrow, 0, off, KPAD);
        v16h a1 = ldsFrag(qb, mrow, 1, off, KPAD);
        v8f acc[4];
        for (int t = 0; t < 4; ++t) {
            v16h b0 = ldsFrag(kb, t * 16 + laneLo, 0, off, KPAD);
            v16h b1 = ldsFrag(kb, t * 16 + laneLo, 1, off, KPAD);
            v8f c0 = {};
            c0 = __builtin_amdgcn_wmma_f32_16x16x32_f16(false, a0, false, b0, (short)0, c0, false, false);
            c0 = __builtin_amdgcn_wmma_f32_16x16x32_f16(false, a1, false, b1, (short)0, c0, false, false);
            acc[t] = c0;
        }

        // ---- + relative-position bias + window mask ----
        for (int t = 0; t < 4; ++t) {
            int kp = t * 16 + laneLo;
            int kh = kp >> 3, kwc = kp & 7;
            for (int r = 0; r < 8; ++r) {
                int qp = strip * 16 + r + off;   // off doubles as +8 row shift for laneHi
                int qh = qp >> 3, qwc = qp & 7;
                int bidx = (qh - kh + 7) * 15 + (qwc - kwc + 7);
                acc[t][r] += bias_table[bidx * NH + head] + mk[qp * NP + kp];
            }
        }

        // ---- softmax over k (rows live across the 16 lanes of a half-wave) ----
        for (int r = 0; r < 8; ++r) {
            float mx = fmaxf(fmaxf(acc[0][r], acc[1][r]), fmaxf(acc[2][r], acc[3][r]));
            mx = fmaxf(mx, __shfl_xor(mx, 1, 32));
            mx = fmaxf(mx, __shfl_xor(mx, 2, 32));
            mx = fmaxf(mx, __shfl_xor(mx, 4, 32));
            mx = fmaxf(mx, __shfl_xor(mx, 8, 32));
            float sum = 0.f;
            for (int t = 0; t < 4; ++t) {
                float e = __expf(acc[t][r] - mx);
                acc[t][r] = e;
                sum += e;
            }
            sum += __shfl_xor(sum, 1, 32);
            sum += __shfl_xor(sum, 2, 32);
            sum += __shfl_xor(sum, 4, 32);
            sum += __shfl_xor(sum, 8, 32);
            float inv = 1.f / sum;
            for (int t = 0; t < 4; ++t) acc[t][r] *= inv;
        }

        // ---- write probs over this wave's (private) q strip region in LDS ----
        for (int t = 0; t < 4; ++t)
            for (int r = 0; r < 8; ++r) {
                int rr = strip * 16 + r + off;
                q_t[grp][rr * KPAD + t * 16 + laneLo] = (_Float16)acc[t][r];
            }

        // ---- out = P @ V : A from prob strip, B from vT ----
        v16h pa0 = ldsFrag(qb, mrow, 0, off, KPAD);
        v16h pa1 = ldsFrag(qb, mrow, 1, off, KPAD);
        for (int t = 0; t < 4; ++t) {           // t = output s index (16 cols each)
            v16h b0 = ldsFrag(vb, t * 16 + laneLo, 0, off, KPAD);
            v16h b1 = ldsFrag(vb, t * 16 + laneLo, 1, off, KPAD);
            v8f o = {};
            o = __builtin_amdgcn_wmma_f32_16x16x32_f16(false, pa0, false, b0, (short)0, o, false, false);
            o = __builtin_amdgcn_wmma_f32_16x16x32_f16(false, pa1, false, b1, (short)0, o, false, false);
            // store pre-projection X in reverse-grouped row order
            for (int r = 0; r < 8; ++r) {
                int qp = strip * 16 + r + off;
                xw[rowOf(qp, t) * ED + head * HD + laneLo] = o[r];
            }
        }
        __syncthreads();
    }
}

__global__ __launch_bounds__(128)
void dwa_proj_kernel(const float* __restrict__ proj_w, const float* __restrict__ proj_b,
                     float* __restrict__ io)
{
    __shared__ __align__(16) _Float16 w_s[ED * WPAD];   // proj_w rows (j-major): B(K=i,N=j)=W[j][i]
    __shared__ __align__(16) _Float16 x_s[64 * WPAD];
    __shared__ float pb_s[ED];

    const int w   = blockIdx.x >> 2;
    const int rb  = (blockIdx.x & 3) * 64;
    const int tid = threadIdx.x;
    float* rows = io + (size_t)w * (ROWS * ED) + (size_t)rb * ED;

    for (int idx = tid; idx < ED * (ED / 4); idx += 128) {   // 96x24 float4
        int r  = idx / 24;
        int c4 = (idx % 24) * 4;
        float4 wv = *(const float4*)(proj_w + r * ED + c4);
        v4h hh;
        hh[0] = (_Float16)wv.x;  hh[1] = (_Float16)wv.y;
        hh[2] = (_Float16)wv.z;  hh[3] = (_Float16)wv.w;
        *(v4h*)(&w_s[r * WPAD + c4]) = hh;
    }
    for (int idx = tid; idx < 64 * (ED / 4); idx += 128) {
        int r  = idx / 24;
        int c4 = (idx % 24) * 4;
        float4 xv = *(const float4*)(rows + r * ED + c4);
        v4h hh;
        hh[0] = (_Float16)xv.x;  hh[1] = (_Float16)xv.y;
        hh[2] = (_Float16)xv.z;  hh[3] = (_Float16)xv.w;
        *(v4h*)(&x_s[r * WPAD + c4]) = hh;
    }
    if (tid < ED) pb_s[tid] = proj_b[tid];
    __syncthreads();

    const int lane   = tid & 31;
    const int wave   = tid >> 5;
    const int laneLo = lane & 15;
    const int off    = (lane >> 4) << 3;
    const int mrow   = wave * 16 + laneLo;

    v16h a0 = ldsFrag(x_s, mrow, 0, off, WPAD);
    v16h a1 = ldsFrag(x_s, mrow, 1, off, WPAD);
    v16h a2 = ldsFrag(x_s, mrow, 2, off, WPAD);

    for (int nt = 0; nt < 6; ++nt) {
        v16h b0 = ldsFrag(w_s, nt * 16 + laneLo, 0, off, WPAD);
        v16h b1 = ldsFrag(w_s, nt * 16 + laneLo, 1, off, WPAD);
        v16h b2 = ldsFrag(w_s, nt * 16 + laneLo, 2, off, WPAD);
        v8f o = {};
        o = __builtin_amdgcn_wmma_f32_16x16x32_f16(false, a0, false, b0, (short)0, o, false, false);
        o = __builtin_amdgcn_wmma_f32_16x16x32_f16(false, a1, false, b1, (short)0, o, false, false);
        o = __builtin_amdgcn_wmma_f32_16x16x32_f16(false, a2, false, b2, (short)0, o, false, false);
        float pb = pb_s[nt * 16 + laneLo];
        for (int r = 0; r < 8; ++r) {
            int m = wave * 16 + r + off;
            rows[m * ED + nt * 16 + laneLo] = o[r] + pb;
        }
    }
}

extern "C" void kernel_launch(void* const* d_in, const int* in_sizes, int n_in,
                              void* d_out, int out_size, void* d_ws, size_t ws_size,
                              hipStream_t stream) {
    const float* q     = (const float*)d_in[0];
    const float* k     = (const float*)d_in[1];
    const float* v     = (const float*)d_in[2];
    const float* mask  = (const float*)d_in[3];
    const float* bt    = (const float*)d_in[4];
    const float* gamma = (const float*)d_in[5];
    const float* beta  = (const float*)d_in[6];
    const float* pw    = (const float*)d_in[7];
    const float* pb    = (const float*)d_in[8];
    float* out = (float*)d_out;

    const int nw = in_sizes[0] / (ROWS * ED);   // 2048

    dwa_attn_kernel<<<nw, 256, 0, stream>>>(q, k, v, mask, bt, gamma, beta, out);
    dwa_proj_kernel<<<nw * 4, 128, 0, stream>>>(pw, pb, out);
}
